// LinearAttentionBlock_22239340658977
// MI455X (gfx1250) — matrix operand order repaired and checked by reading, not compiled, not run on hardware
//
#include <hip/hip_runtime.h>
#include <hip/hip_bf16.h>

// CDNA5 (gfx1250) wave32 WMMA implementation of the linear-attention block.
// GEMM activations are stored pixel-major [N, K] so every WMMA A *and* B
// fragment is a pair of contiguous 16B LDS reads (ds_load_b128), and bf16
// epilogues feeding the next GEMM are 16B vector stores per lane.

typedef __attribute__((ext_vector_type(16))) __bf16 v16bf;
typedef __attribute__((ext_vector_type(8)))  __bf16 v8bf;
typedef __attribute__((ext_vector_type(8)))  float  v8f;

static constexpr int HEADS = 12;
static constexpr int KD    = 16;
static constexpr int NB    = 16;     // batch
static constexpr int CC    = 384;    // channels
static constexpr int NN    = 4096;   // H*W
static constexpr int SS    = 192;    // HEADS*KD
static constexpr int MPROJ = 576;    // 3*S
static constexpr int KPROJ = 386;    // C+2
static constexpr int KPPAD = 416;    // padded to multiple of 32

__device__ __forceinline__ float hswish(float x) {
  return x * fminf(fmaxf(x + 3.0f, 0.0f), 6.0f) * (1.0f / 6.0f);
}

// ---------------------------------------------------------------------------
// Weight fold: dst[m, kp] = bf16(g[m] * W[m, k]) for k < K, zero-pad beyond.
// ---------------------------------------------------------------------------
__global__ void fold_weight(const float* __restrict__ W, const float* __restrict__ g,
                            __bf16* __restrict__ dst, int M, int K, int Kp) {
  int idx = blockIdx.x * blockDim.x + threadIdx.x;
  if (idx >= M * Kp) return;
  int m = idx / Kp, k = idx % Kp;
  float v = (k < K) ? g[m] * W[m * K + k] : 0.0f;
  dst[idx] = (__bf16)v;
}

// ---------------------------------------------------------------------------
// Transpose x [B,384,4096] f32 -> xe [B,4096,416] bf16 (channels 0..383).
// LDS-tiled 32x32 so both global read and write are coalesced.
// ---------------------------------------------------------------------------
__global__ __launch_bounds__(256) void xemb_transpose(
    const float* __restrict__ x, __bf16* __restrict__ xe) {
  __shared__ float tile[32][33];
  const int b  = blockIdx.z;
  const int gk = blockIdx.y * 32;
  const int gn = blockIdx.x * 32;
  const int tx = threadIdx.x & 31;
  const int ty = threadIdx.x >> 5;  // 0..7
#pragma unroll
  for (int i = 0; i < 4; ++i)
    tile[ty + i * 8][tx] = x[((size_t)b * CC + gk + ty + i * 8) * NN + gn + tx];
  __syncthreads();
#pragma unroll
  for (int i = 0; i < 4; ++i)
    xe[((size_t)b * NN + gn + ty + i * 8) * KPPAD + gk + tx] =
        (__bf16)tile[tx][ty + i * 8];
}

// Fill xe channels 384..415: 2 positional channels + 30 zero pad.
__global__ void xemb_pos(const float* __restrict__ rnd, const float* __restrict__ es_ptr,
                         __bf16* __restrict__ xe) {
  int idx = blockIdx.x * blockDim.x + threadIdx.x;   // NB*NN*32 = 2^21
  if (idx >= NB * NN * 32) return;
  int k = idx & 31;
  int n = (idx >> 5) & (NN - 1);
  int b = idx >> 17;                                  // NN*32 = 2^17
  float v = 0.0f;
  float e = expf(*es_ptr);
  if (k == 0)      v = e * ((float)(n & 63) * 0.001f + rnd[b * 2 + 0]);  // xx
  else if (k == 1) v = e * ((float)(n >> 6) * 0.001f + rnd[b * 2 + 1]);  // yy
  xe[((size_t)b * NN + n) * KPPAD + CC + k] = (__bf16)v;
}

// ---------------------------------------------------------------------------
// Generic bf16 WMMA GEMM: O[b] = W(MxKp) @ X[b](N x Kp, pixel-major) + bias.
// Block 256 threads (8 waves). Tile: 64(M) x 128(N), K-step 32.
// Wave (wm 0..3, wn 0..1) computes 16(M) x 64(N) via 4 accumulators.
// mode 0: bf16(relu(t))            -> outB [M,N]   (proj, channel-major)
// mode 1: f32(resid + rz*relu(t))  -> outF [M,N]
// mode 2: bf16(hard_swish(t))      -> outB [N,M]   (feeds next GEMM)
// mode 3: f32(resid + rz*t)        -> outF [M,N]
// ---------------------------------------------------------------------------
__global__ __launch_bounds__(256) void gemm_bf16_wmma(
    const __bf16* __restrict__ W, const __bf16* __restrict__ X,
    const float* __restrict__ bias, int M, int Kp, int N, int mode,
    float* __restrict__ outF, __bf16* __restrict__ outB,
    const float* __restrict__ resid, const float* __restrict__ rz) {
  __shared__ __bf16 lsA[64][40];    // 64 M-rows x 32 K (padded)
  __shared__ __bf16 lsB[128][40];   // 128 N-rows x 32 K (padded)

  const int tid  = threadIdx.x;
  const int lane = tid & 31;
  const int wv   = tid >> 5;
  const int wm   = wv & 3;
  const int wn   = wv >> 2;
  const int hi   = lane >> 4;
  const int lm   = lane & 15;
  const int n0   = blockIdx.x * 128;
  const int m0   = blockIdx.y * 64;
  const int b    = blockIdx.z;
  const __bf16* Xb = X + (size_t)b * N * Kp;

  v8f acc[4] = {};

  for (int k0 = 0; k0 < Kp; k0 += 32) {
    // Stage A (64x32): 8 contiguous bf16 (16B) per thread.
    {
      int row = tid >> 2, col = (tid & 3) * 8;
      const __bf16* src = W + (size_t)(m0 + row) * Kp + k0 + col;
#pragma unroll
      for (int i = 0; i < 8; ++i) lsA[row][col + i] = src[i];
    }
    // Stage B (128 pixels x 32 k): 16 contiguous bf16 (32B) per thread.
    {
      int row = tid >> 1, col = (tid & 1) * 16;
      const __bf16* src = Xb + (size_t)(n0 + row) * Kp + k0 + col;
#pragma unroll
      for (int i = 0; i < 16; ++i) lsB[row][col + i] = src[i];
      if (k0 + 32 < Kp)
        __builtin_prefetch(Xb + (size_t)(n0 + row) * Kp + k0 + 32 + col, 0, 1);
    }
    __syncthreads();

    // A fragment: K = j + 16*hi -> one contiguous 32B run (matches B).
    v16bf af;
#pragma unroll
    for (int j = 0; j < 16; ++j) {
      int kk = 16 * hi + j;
      af[j] = lsA[wm * 16 + lm][kk];
    }
#pragma unroll
    for (int t = 0; t < 4; ++t) {
      // B fragment: lane = pixel n, K = j + 16*hi -> one contiguous 32B run.
      v16bf bfr;
#pragma unroll
      for (int j = 0; j < 16; ++j)
        bfr[j] = lsB[wn * 64 + t * 16 + lm][16 * hi + j];
      acc[t] = __builtin_amdgcn_wmma_f32_16x16x32_bf16(
          false, af, false, bfr, (short)0, acc[t], false, false);
    }
    __syncthreads();
  }

  const float rzv = (rz != nullptr) ? *rz : 0.0f;
  const int mbase = m0 + wm * 16 + 8 * hi;   // D layout: M = r + 8*hi
  if (mode == 2) {
#pragma unroll
    for (int t = 0; t < 4; ++t) {
      int n = n0 + wn * 64 + t * 16 + lm;
      v8bf o;
#pragma unroll
      for (int r = 0; r < 8; ++r)
        o[r] = (__bf16)hswish(acc[t][r] + bias[mbase + r]);
      *(v8bf*)(outB + ((size_t)b * N + n) * M + mbase) = o;   // [N,M], 16B store
    }
  } else {
#pragma unroll
    for (int t = 0; t < 4; ++t) {
      int n = n0 + wn * 64 + t * 16 + lm;
#pragma unroll
      for (int r = 0; r < 8; ++r) {
        int m = mbase + r;
        float v = acc[t][r] + bias[m];
        size_t oi = (size_t)b * M * N + (size_t)m * N + n;
        if (mode == 0)      outB[oi] = (__bf16)fmaxf(v, 0.0f);
        else if (mode == 1) outF[oi] = resid[oi] + rzv * fmaxf(v, 0.0f);
        else                outF[oi] = resid[oi] + rzv * v;
      }
    }
  }
}

// ---------------------------------------------------------------------------
// ktv[d][e] = sum_n k[n][d]*v[n][e],  ksum[d] = sum_n k[n][d] per (b,head).
// proj is [C,N]: A = k^T rows and B = v columns are n-contiguous -> direct
// global b128 fragment loads, no LDS staging.
// ---------------------------------------------------------------------------
__global__ __launch_bounds__(256) void ktv_wmma(
    const __bf16* __restrict__ proj, float* __restrict__ ktv,
    float* __restrict__ ksum) {
  __shared__ float red[8][256];
  __shared__ float kred[256];
  const int tid  = threadIdx.x;
  const int lane = tid & 31;
  const int wv   = tid >> 5;
  const int hi   = lane >> 4;
  const int lm   = lane & 15;
  const int bh   = blockIdx.x;
  const int b    = bh / HEADS, h = bh % HEADS;
  const __bf16* pb   = proj + (size_t)b * MPROJ * NN;
  const __bf16* krow = pb + (size_t)(SS + h * KD + lm) * NN;
  const __bf16* vrow = pb + (size_t)(2 * SS + h * KD + lm) * NN;

  v8f acc = {};
  for (int chunk = wv; chunk < NN / 32; chunk += 8) {
    int n0 = chunk * 32;
    v16bf af, bfr;
#pragma unroll
    for (int j = 0; j < 16; ++j) {
      int kk = 16 * hi + j;
      af[j] = krow[n0 + kk];
    }
#pragma unroll
    for (int j = 0; j < 16; ++j) bfr[j] = vrow[n0 + 16 * hi + j];
    acc = __builtin_amdgcn_wmma_f32_16x16x32_bf16(
        false, af, false, bfr, (short)0, acc, false, false);
  }
#pragma unroll
  for (int r = 0; r < 8; ++r) red[r][wv * 32 + lane] = acc[r];

  {  // ksum partials
    int d = tid & 15, seg = tid >> 4;
    const __bf16* kr = pb + (size_t)(SS + h * KD + d) * NN + seg * 256;
    float s = 0.0f;
    for (int i = 0; i < 256; ++i) s += (float)kr[i];
    kred[tid] = s;
  }
  __syncthreads();
  {
    int r = tid >> 5, ln = tid & 31;
    float s = 0.0f;
#pragma unroll
    for (int w = 0; w < 8; ++w) s += red[r][w * 32 + ln];
    int m = r + 8 * (ln >> 4), e = ln & 15;
    ktv[(size_t)bh * 256 + m * 16 + e] = s;
  }
  if (tid < 16) {
    float s = 0.0f;
#pragma unroll
    for (int seg = 0; seg < 16; ++seg) s += kred[seg * 16 + tid];
    ksum[bh * 16 + tid] = s;
  }
}

// ---------------------------------------------------------------------------
// attn[b, n, h*16+e] = (sum_d q[n][d]*ktv[d][e]) / max(q.ksum, 1e-4)
// Output is pixel-major [N, 192]; 2x16B vector store per thread.
// ---------------------------------------------------------------------------
__global__ __launch_bounds__(256) void attn_out(
    const __bf16* __restrict__ proj, const float* __restrict__ ktv,
    const float* __restrict__ ksum, __bf16* __restrict__ attn) {
  __shared__ float kt[256];
  __shared__ float ks[16];
  const int tid = threadIdx.x;
  const int bh  = blockIdx.y;
  const int b   = bh / HEADS, h = bh % HEADS;
  kt[tid] = ktv[(size_t)bh * 256 + tid];
  if (tid < 16) ks[tid] = ksum[bh * 16 + tid];
  __syncthreads();
  const int n = blockIdx.x * 256 + tid;
  const __bf16* pb = proj + (size_t)b * MPROJ * NN;
  float q[16], denom = 0.0f;
#pragma unroll
  for (int d = 0; d < 16; ++d) {
    q[d] = (float)pb[(size_t)(h * KD + d) * NN + n];
    denom += q[d] * ks[d];
  }
  const float inv = 1.0f / fmaxf(denom, 1e-4f);
  v8bf o0, o1;
#pragma unroll
  for (int e = 0; e < 16; ++e) {
    float v = 0.0f;
#pragma unroll
    for (int d = 0; d < 16; ++d) v += q[d] * kt[d * 16 + e];
    __bf16 ov = (__bf16)(v * inv);
    if (e < 8) o0[e] = ov; else o1[e - 8] = ov;
  }
  __bf16* dst = attn + ((size_t)b * NN + n) * SS + h * KD;
  *(v8bf*)dst = o0;
  *(v8bf*)(dst + 8) = o1;
}

// ---------------------------------------------------------------------------
// Depthwise 5x5 SAME conv on y [C,N] f32 + affine + hard_swish -> z [N,C] bf16.
// Block = 16 channels x 16x16 spatial tile; channel-fast thread mapping so
// the pixel-major bf16 stores are coalesced across lanes.
// ---------------------------------------------------------------------------
__global__ __launch_bounds__(256) void dwconv_hswish(
    const float* __restrict__ y, const float* __restrict__ dww,
    const float* __restrict__ dwg, const float* __restrict__ dwb,
    __bf16* __restrict__ z) {
  __shared__ float tile[16][20][21];
  __shared__ float wk[16][25];
  __shared__ float gg[16], bb[16];
  const int tid = threadIdx.x;
  const int b   = blockIdx.z;
  const int c0  = blockIdx.y * 16;
  const int gx0 = (blockIdx.x & 3) * 16, gy0 = (blockIdx.x >> 2) * 16;
  if (tid < 16 * 25) wk[tid / 25][tid % 25] = dww[(c0 + tid / 25) * 25 + tid % 25];
  if (tid < 16) { gg[tid] = dwg[c0 + tid]; bb[tid] = dwb[c0 + tid]; }
  for (int idx = tid; idx < 16 * 400; idx += 256) {
    int c = idx / 400, p = idx % 400;
    int ly = p / 20, lx = p % 20;
    int gy = gy0 + ly - 2, gx = gx0 + lx - 2;
    float v = 0.0f;
    if (gy >= 0 && gy < 64 && gx >= 0 && gx < 64)
      v = y[((size_t)b * CC + c0 + c) * NN + gy * 64 + gx];
    tile[c][ly][lx] = v;
  }
  __syncthreads();
  const int c  = tid & 15;       // channel-fast: coalesced [N,C] stores
  const int py = tid >> 4;       // row of the 16x16 tile
#pragma unroll 4
  for (int px = 0; px < 16; ++px) {
    float acc = 0.0f;
#pragma unroll
    for (int dy = 0; dy < 5; ++dy)
#pragma unroll
      for (int dx = 0; dx < 5; ++dx)
        acc += tile[c][py + dy][px + dx] * wk[c][dy * 5 + dx];
    float t = acc * gg[c] + bb[c];
    int n = (gy0 + py) * 64 + gx0 + px;
    z[((size_t)b * NN + n) * CC + c0 + c] = (__bf16)hswish(t);
  }
}

// ---------------------------------------------------------------------------
extern "C" void kernel_launch(void* const* d_in, const int* in_sizes, int n_in,
                              void* d_out, int out_size, void* d_ws, size_t ws_size,
                              hipStream_t stream) {
  const float* x       = (const float*)d_in[0];
  const float* rnd     = (const float*)d_in[1];
  const float* proj_w  = (const float*)d_in[2];
  const float* proj_g  = (const float*)d_in[3];
  const float* proj_b  = (const float*)d_in[4];
  const float* vp2_w   = (const float*)d_in[5];
  const float* vp2_g   = (const float*)d_in[6];
  const float* vp2_b   = (const float*)d_in[7];
  const float* dw_w    = (const float*)d_in[8];
  const float* dw_g    = (const float*)d_in[9];
  const float* dw_b    = (const float*)d_in[10];
  const float* pw1_w   = (const float*)d_in[11];
  const float* pw1_g   = (const float*)d_in[12];
  const float* pw1_b   = (const float*)d_in[13];
  const float* pw2_w   = (const float*)d_in[14];
  const float* out_g   = (const float*)d_in[15];
  const float* out_b   = (const float*)d_in[16];
  const float* rz_attn = (const float*)d_in[17];
  const float* rz_ffn  = (const float*)d_in[18];
  const float* es      = (const float*)d_in[19];

  char* ws = (char*)d_ws;
  size_t off = 0;
  auto alloc = [&](size_t bytes) {
    void* p = ws + off;
    off = (off + bytes + 255) & ~(size_t)255;
    return p;
  };
  __bf16* xe   = (__bf16*)alloc((size_t)NB * NN * KPPAD * 2);   // [B,N,416] 54.5 MB
  __bf16* proj = (__bf16*)alloc((size_t)NB * MPROJ * NN * 2);   // [B,576,N] 75.5 MB
  float*  ybuf = (float*) alloc((size_t)NB * CC * NN * 4);      // [B,384,N] 100.7 MB
  __bf16* hbuf = (__bf16*)alloc((size_t)NB * NN * 4 * CC * 2);  // [B,N,1536] 201.3 MB
  float*  ktvb = (float*) alloc((size_t)NB * HEADS * 256 * 4);
  float*  ksb  = (float*) alloc((size_t)NB * HEADS * 16 * 4);
  __bf16* wpj  = (__bf16*)alloc((size_t)MPROJ * KPPAD * 2);
  __bf16* wv2  = (__bf16*)alloc((size_t)CC * SS * 2);
  __bf16* wp1  = (__bf16*)alloc((size_t)4 * CC * CC * 2);
  __bf16* wp2  = (__bf16*)alloc((size_t)CC * 4 * CC * 2);
  __bf16* attn = xe;    // reuse: [B,N,192] 25.2 MB (xe last read: proj GEMM)
  __bf16* zbuf = proj;  // reuse: [B,N,384] 50.3 MB (proj last read: attn_out)

  dim3 blk256(256);
  fold_weight<<<(MPROJ * KPPAD + 255) / 256, blk256, 0, stream>>>(proj_w, proj_g, wpj, MPROJ, KPROJ, KPPAD);
  fold_weight<<<(CC * SS + 255) / 256, blk256, 0, stream>>>(vp2_w, vp2_g, wv2, CC, SS, SS);
  fold_weight<<<(4 * CC * CC + 255) / 256, blk256, 0, stream>>>(pw1_w, pw1_g, wp1, 4 * CC, CC, CC);
  fold_weight<<<(CC * 4 * CC + 255) / 256, blk256, 0, stream>>>(pw2_w, out_g, wp2, CC, 4 * CC, 4 * CC);

  xemb_transpose<<<dim3(NN / 32, CC / 32, NB), blk256, 0, stream>>>(x, xe);
  xemb_pos<<<(NB * NN * 32 + 255) / 256, blk256, 0, stream>>>(rnd, es, xe);

  // proj = relu((g*W) x_emb + b) -> bf16 [B, 576, N] (channel-major for attn)
  gemm_bf16_wmma<<<dim3(NN / 128, MPROJ / 64, NB), blk256, 0, stream>>>(
      wpj, xe, proj_b, MPROJ, KPPAD, NN, 0, nullptr, proj, nullptr, nullptr);

  ktv_wmma<<<NB * HEADS, blk256, 0, stream>>>(proj, ktvb, ksb);
  attn_out<<<dim3(NN / 256, NB * HEADS), blk256, 0, stream>>>(proj, ktvb, ksb, attn);

  // y = x + rz_attn * relu((g*W) attn + b) -> f32 [B, 384, N]
  gemm_bf16_wmma<<<dim3(NN / 128, CC / 64, NB), blk256, 0, stream>>>(
      wv2, attn, vp2_b, CC, SS, NN, 1, ybuf, nullptr, x, rz_attn);

  dwconv_hswish<<<dim3(16, CC / 16, NB), blk256, 0, stream>>>(ybuf, dw_w, dw_g, dw_b, zbuf);

  // h = hswish((g*W) z + b) -> bf16 [B, N, 1536]
  gemm_bf16_wmma<<<dim3(NN / 128, (4 * CC) / 64, NB), blk256, 0, stream>>>(
      wp1, zbuf, pw1_b, 4 * CC, CC, NN, 2, nullptr, hbuf, nullptr, nullptr);

  // out = y + rz_ffn * ((out_g*W) h + out_b) -> f32 d_out [B, 384, N]
  gemm_bf16_wmma<<<dim3(NN / 128, CC / 64, NB), blk256, 0, stream>>>(
      wp2, hbuf, out_b, CC, 4 * CC, NN, 3, (float*)d_out, nullptr, ybuf, rz_ffn);
}
